// MLP_52759378264652
// MI455X (gfx1250) — matrix operand order repaired and checked by reading, not compiled, not run on hardware
//
#include <hip/hip_runtime.h>

// CDNA5 / gfx1250: wave32, WMMA f32_16x16x32_f16.
// Four independent 4x4x16 GEMMs packed block-diagonally into one WMMA:
//   group g in {0,1,2,3}: A rows 4g..4g+3, K-block {0-3,4-7,16-19,20-23}.
// With these K-blocks, D -> next-layer B is a pure per-lane f16 pack
// (no cross-lane permutes), with this code's contiguous K numbering k = 16*(lane/16) + elem.
// TWO independent chains per wave (128 samples): one shared A-build feeds two
// WMMAs; each chain's VALU work fills the other's WMMA->VALU hazard slots.
// Sigmoid via hardware V_TANH_F32: sigmoid(t) = 0.5 + 0.5*tanh(t/2); the 1/2
// is folded into the WMMA weights (A = 0.5*W) and the bias pk_fma constant,
// and the final affine (0.5*th + 0.5) is done in packed f16 (v_pk_fma_f16)
// after the cvt_pk, halving the tail op count. The head consumes raw tanh
// values with algebraically folded weights.

typedef __attribute__((ext_vector_type(16))) _Float16 v16h;
typedef __attribute__((ext_vector_type(8)))  float    v8f;
typedef __attribute__((ext_vector_type(2)))  _Float16 v2h;
typedef __attribute__((ext_vector_type(2)))  __fp16   v2hf;   // builtin's return type

union Pk2 { v2hf hf; v2h h; unsigned u; };

static __device__ __forceinline__ unsigned pk2u(float a, float b) {
  Pk2 p; p.hf = __builtin_amdgcn_cvt_pkrtz(a, b); return p.u;  // v_cvt_pk_rtz_f16_f32
}
static __device__ __forceinline__ void put2(v16h& v, int i, unsigned u) {
  Pk2 p; p.u = u; v[i] = p.h[0]; v[i + 1] = p.h[1];
}

static __device__ __forceinline__ float tanh_hw(float u) {
#if __has_builtin(__builtin_amdgcn_tanhf)
  return __builtin_amdgcn_tanhf(u);
#else
  float th;
  asm("v_tanh_f32 %0, %1" : "=v"(th) : "v"(u));
  return th;
#endif
}

// packed-f16 affine: h2 = 0.5*th2 + 0.5 per half  (v_pk_fma_f16, inline 0.5)
static __device__ __forceinline__ unsigned affine_h2(unsigned th2) {
  const v2h c = { (_Float16)0.5f, (_Float16)0.5f };
  Pk2 p; p.u = th2;
  Pk2 r; r.h = __builtin_elementwise_fma(p.h, c, c);
  return r.u;
}

__global__ __launch_bounds__(256) void mlp_chain_wmma(
    const float* __restrict__ x,      // [B,2]
    const float* __restrict__ W_in,   // [4,2]
    const float* __restrict__ b_in,   // [4]
    const float* __restrict__ W_mid,  // [L,4,4]
    const float* __restrict__ b_mid,  // [L,4]
    const float* __restrict__ W_out,  // [1,4]
    const float* __restrict__ b_out,  // [1]
    float* __restrict__ out,          // [B,1]
    int B, int L)
{
  const int lane = (int)(threadIdx.x & 31);
  const int wave = (int)((blockIdx.x * blockDim.x + threadIdx.x) >> 5);
  const int base = wave * 128;                      // 128 samples per wave, 2 chains

  // lane -> samples. Chain 0: groups g0/g2 cols (idx0) + g1/g3 cols (idx1);
  // chain 1 is the same pattern shifted by 64.
  const int idx0 = base + ((lane < 16) ? lane : lane + 16);
  const int idx1 = idx0 + 16;
  const int idx2 = idx0 + 64;
  const int idx3 = idx1 + 64;
  const int i0c = (idx0 < B) ? idx0 : (B - 1);
  const int i1c = (idx1 < B) ? idx1 : (B - 1);
  const int i2c = (idx2 < B) ? idx2 : (B - 1);
  const int i3c = (idx3 < B) ? idx3 : (B - 1);

  // A-matrix row-block ownership: lanes 0..7 carry groups 0/1 (K 0-7),
  // lanes 24..31 carry groups 2/3 (K 16-23); row (lane&3) of W per lane.
  const int g    = (lane < 8) ? (lane >> 2) : ((lane >= 24) ? ((lane >> 2) - 4) : -1);
  const int wrow = lane & 3;

  // ---------------- stem: h = relu(x @ W_in^T + b_in), fp32 scalar ----------------
  const float2 xa = ((const float2*)x)[i0c];
  const float2 xb = ((const float2*)x)[i1c];
  const float2 xc = ((const float2*)x)[i2c];
  const float2 xd = ((const float2*)x)[i3c];
  float ha[4], hb[4], hc[4], hd[4];
#pragma unroll
  for (int j = 0; j < 4; ++j) {
    const float w0 = W_in[2 * j], w1 = W_in[2 * j + 1], bj = b_in[j];
    ha[j] = fmaxf(0.f, fmaf(xa.x, w0, fmaf(xa.y, w1, bj)));
    hb[j] = fmaxf(0.f, fmaf(xb.x, w0, fmaf(xb.y, w1, bj)));
    hc[j] = fmaxf(0.f, fmaf(xc.x, w0, fmaf(xc.y, w1, bj)));
    hd[j] = fmaxf(0.f, fmaf(xd.x, w0, fmaf(xd.y, w1, bj)));
  }

  // B matrices: elems 0-3 = h(idxEven) (K 0-3 lo / 16-19 hi), elems 4-7 = h(idxOdd),
  // elems 8-15 (K 8-15 lo / 24-31 hi) stay zero (matching A zeros).
  v16h bm0 = {}, bm1 = {};
  put2(bm0, 0, pk2u(ha[0], ha[1]));
  put2(bm0, 2, pk2u(ha[2], ha[3]));
  put2(bm0, 4, pk2u(hb[0], hb[1]));
  put2(bm0, 6, pk2u(hb[2], hb[3]));
  put2(bm1, 0, pk2u(hc[0], hc[1]));
  put2(bm1, 2, pk2u(hc[2], hc[3]));
  put2(bm1, 4, pk2u(hd[0], hd[1]));
  put2(bm1, 6, pk2u(hd[2], hd[3]));

  const v8f czero = {};
  // th arrays hold raw tanh values; seed with 2h-1 so the folded head
  // (out = sum(0.5*w*th) + b + 0.5*sum(w)) is exact even if L == 0.
  float th0[8], th1[8];
#pragma unroll
  for (int j = 0; j < 4; ++j) {
    th0[j]     = fmaf(2.f, ha[j], -1.f);
    th0[j + 4] = fmaf(2.f, hb[j], -1.f);
    th1[j]     = fmaf(2.f, hc[j], -1.f);
    th1[j + 4] = fmaf(2.f, hd[j], -1.f);
  }

  // Software-pipelined per-lane W row load (16B, L0-resident: whole W_mid is 6.3KB)
  float4 wr = ((const float4*)W_mid)[0 * 4 + wrow];

  for (int l = 0; l < L; ++l) {
    const int    ln  = (l + 1 < L) ? (l + 1) : l;
    const float4 wrn = ((const float4*)W_mid)[ln * 4 + wrow];  // prefetch next layer

    // Build A = blockdiag(0.5*W_l) (folds the tanh half-argument).
    // 0.5 is an inline constant; shared by both chains (128 samples).
    const unsigned t01 = pk2u(wr.x * 0.5f, wr.y * 0.5f);
    const unsigned t23 = pk2u(wr.z * 0.5f, wr.w * 0.5f);
    v16h amat = {};
    put2(amat, 0,  (g == 0 || g == 2) ? t01 : 0u);   // A V0: K 0,1 / 16,17 (hi)
    put2(amat, 2,  (g == 0 || g == 2) ? t23 : 0u);   // A V1: K 2,3 / 18,19 (hi)
    put2(amat, 4,  (g == 1 || g == 3) ? t01 : 0u);   // A V2: K 4,5 / 20,21 (hi)
    put2(amat, 6,  (g == 1 || g == 3) ? t23 : 0u);   // A V3: K 6,7 / 22,23 (hi)
    put2(amat, 8,  0u);
    put2(amat, 10, 0u);
    put2(amat, 12, 0u);
    put2(amat, 14, 0u);

    // Two WMMAs, disjoint D/B registers: no D->A/B hazard between them.
    const v8f d0 = __builtin_amdgcn_wmma_f32_16x16x32_f16(
        false, amat, false, bm0, (short)0, czero, false, false);
    const v8f d1 = __builtin_amdgcn_wmma_f32_16x16x32_f16(
        false, amat, false, bm1, (short)0, czero, false, false);

    // Bias is uniform per D-VGPR: d[r] is hidden unit (r&3) of some group.
    // u = d + 0.5*b (pk_fma, inline 0.5), th = tanh(u) (hardware TRANS).
    const float bb0 = b_mid[l * 4 + 0] * 0.5f;
    const float bb1 = b_mid[l * 4 + 1] * 0.5f;
    const float bb2 = b_mid[l * 4 + 2] * 0.5f;
    const float bb3 = b_mid[l * 4 + 3] * 0.5f;

    th0[0] = tanh_hw(d0[0] + bb0);  th0[1] = tanh_hw(d0[1] + bb1);
    th0[2] = tanh_hw(d0[2] + bb2);  th0[3] = tanh_hw(d0[3] + bb3);
    th0[4] = tanh_hw(d0[4] + bb0);  th0[5] = tanh_hw(d0[5] + bb1);
    th0[6] = tanh_hw(d0[6] + bb2);  th0[7] = tanh_hw(d0[7] + bb3);
    th1[0] = tanh_hw(d1[0] + bb0);  th1[1] = tanh_hw(d1[1] + bb1);
    th1[2] = tanh_hw(d1[2] + bb2);  th1[3] = tanh_hw(d1[3] + bb3);
    th1[4] = tanh_hw(d1[4] + bb0);  th1[5] = tanh_hw(d1[5] + bb1);
    th1[6] = tanh_hw(d1[6] + bb2);  th1[7] = tanh_hw(d1[7] + bb3);

    // D -> B: pack tanh pairs (cvt_pk), then h = 0.5*th + 0.5 in packed f16
    // (v_pk_fma_f16) — 2 values per instruction, no cross-lane movement.
    put2(bm0, 0, affine_h2(pk2u(th0[0], th0[1])));
    put2(bm0, 2, affine_h2(pk2u(th0[2], th0[3])));
    put2(bm0, 4, affine_h2(pk2u(th0[4], th0[5])));
    put2(bm0, 6, affine_h2(pk2u(th0[6], th0[7])));
    put2(bm1, 0, affine_h2(pk2u(th1[0], th1[1])));
    put2(bm1, 2, affine_h2(pk2u(th1[2], th1[3])));
    put2(bm1, 4, affine_h2(pk2u(th1[4], th1[5])));
    put2(bm1, 6, affine_h2(pk2u(th1[6], th1[7])));

    wr = wrn;
  }

  // -------- head on raw tanh values with folded weights --------
  // out = sum_j w_j*(0.5*th_j + 0.5) + b = sum_j (0.5*w_j)*th_j + (b + 0.5*sum_j w_j)
  const float w0 = 0.5f * W_out[0], w1 = 0.5f * W_out[1];
  const float w2 = 0.5f * W_out[2], w3 = 0.5f * W_out[3];
  const float bo = b_out[0] + w0 + w1 + w2 + w3;
  const float o0 = fmaf(th0[0], w0, fmaf(th0[1], w1, fmaf(th0[2], w2, fmaf(th0[3], w3, bo))));
  const float o1 = fmaf(th0[4], w0, fmaf(th0[5], w1, fmaf(th0[6], w2, fmaf(th0[7], w3, bo))));
  const float o2 = fmaf(th1[0], w0, fmaf(th1[1], w1, fmaf(th1[2], w2, fmaf(th1[3], w3, bo))));
  const float o3 = fmaf(th1[4], w0, fmaf(th1[5], w1, fmaf(th1[6], w2, fmaf(th1[7], w3, bo))));
  if (idx0 < B) out[idx0] = o0;
  if (idx1 < B) out[idx1] = o1;
  if (idx2 < B) out[idx2] = o2;
  if (idx3 < B) out[idx3] = o3;
}

extern "C" void kernel_launch(void* const* d_in, const int* in_sizes, int n_in,
                              void* d_out, int out_size, void* d_ws, size_t ws_size,
                              hipStream_t stream) {
  const float* x     = (const float*)d_in[0];
  const float* W_in  = (const float*)d_in[1];
  const float* b_in  = (const float*)d_in[2];
  const float* W_mid = (const float*)d_in[3];
  const float* b_mid = (const float*)d_in[4];
  const float* W_out = (const float*)d_in[5];
  const float* b_out = (const float*)d_in[6];
  float*       out   = (float*)d_out;

  const int B = in_sizes[0] / 2;    // x is [B,2]
  const int L = in_sizes[3] / 16;   // W_mid is [L,4,4]

  const int threads = (B + 3) / 4;  // 4 samples per thread, 128 per wave
  const int block   = 256;          // 8 waves / block
  const int grid    = (threads + block - 1) / block;

  mlp_chain_wmma<<<grid, block, 0, stream>>>(x, W_in, b_in, W_mid, b_mid,
                                             W_out, b_out, out, B, L);
}
